// GraphAttentionV2_12532714570493
// MI455X (gfx1250) — matrix operand order repaired and checked by reading, not compiled, not run on hardware
//
#include <hip/hip_runtime.h>

typedef __bf16 bf16;
typedef __bf16 v16bf __attribute__((ext_vector_type(16)));
typedef __bf16 v8bf  __attribute__((ext_vector_type(8)));
typedef __bf16 v4bf  __attribute__((ext_vector_type(4)));
typedef __bf16 v2bf  __attribute__((ext_vector_type(2)));
typedef float  v8f   __attribute__((ext_vector_type(8)));
typedef float  v4f   __attribute__((ext_vector_type(4)));
typedef float  v2f   __attribute__((ext_vector_type(2)));

namespace {
constexpr int B_    = 4;
constexpr int S_    = 512;
constexpr int HID_  = 768;
constexpr int NH_   = 12;
constexpr int HD_   = 64;
constexpr int REL_  = 64;
constexpr int BS_   = B_ * S_;        // 2048
constexpr int ROWS_ = B_ * NH_ * S_;  // 24576

constexpr size_t OFF_Q  = 0;
constexpr size_t OFF_K  = OFF_Q  + (size_t)2 * BS_ * HID_;        //  3,145,728
constexpr size_t OFF_V  = OFF_K  + (size_t)2 * BS_ * HID_;        //  6,291,456
constexpr size_t OFF_QR = OFF_V  + (size_t)2 * BS_ * HID_;        //  9,437,184
constexpr size_t OFF_QB = OFF_QR + (size_t)2 * ROWS_ * REL_;      // 12,582,912
constexpr size_t OFF_SC = OFF_QB + (size_t)4 * ROWS_;             // 12,681,216
}

__device__ __forceinline__ bf16 f2bf(float f) {
  unsigned u = __float_as_uint(f);
  u += 0x7FFFu + ((u >> 16) & 1u);                 // round-to-nearest-even
  unsigned short s = (unsigned short)(u >> 16);
  union { unsigned short s; bf16 b; } cv; cv.s = s;
  return cv.b;
}
__device__ __forceinline__ float bf2f(bf16 v) {
  union { bf16 b; unsigned short s; } cv; cv.b = v;
  return __uint_as_float(((unsigned)cv.s) << 16);
}
__device__ __forceinline__ v8f v8f_zero() {
  v8f z;
#pragma unroll
  for (int i = 0; i < 8; ++i) z[i] = 0.0f;
  return z;
}
__device__ __forceinline__ v4bf pack4(v4f f) {
  v4bf r;
  r[0] = f2bf(f[0]); r[1] = f2bf(f[1]); r[2] = f2bf(f[2]); r[3] = f2bf(f[3]);
  return r;
}

// ---- WMMA 16x16x32 bf16 fragment builders (CDNA5 wave32 layouts) ----
// A (16M x 32K): lane rows 0-15; half 0: K {k0+0..7, k0+16..23},
//                half 1: K {k0+8..15, k0+24..31}  -> two b128 LDS loads.
template<int LD>
__device__ __forceinline__ v16bf load_a(const bf16* sm, int lane, int k0) {
  int row = lane & 15, half = lane >> 4;
  const bf16* p = sm + row*LD + k0 + (half << 3);
  union { v16bf v; v8bf h[2]; } u;
  u.h[0] = *(const v8bf*)(p);
  u.h[1] = *(const v8bf*)(p + 16);
  return u.v;
}
// B (32K x 16N), source staged [n][k]: half 0: K {k0+0..7, k0+16..23},
// half 1: K {k0+8..15, k0+24..31} -> two b128 LDS loads.
template<int LD>
__device__ __forceinline__ v16bf load_b(const bf16* sm, int lane, int k0, int n) {
  int half = lane >> 4;
  const bf16* p = sm + n*LD + k0 + (half << 3);
  union { v16bf v; v8bf h[2]; } u;
  u.h[0] = *(const v8bf*)(p);
  u.h[1] = *(const v8bf*)(p + 16);
  return u.v;
}

#define WMMA_BF16(a, b, c) \
  __builtin_amdgcn_wmma_f32_16x16x32_bf16(false, (a), false, (b), (short)0, (c), false, false)

// ============ K1: QKV projection. out[b,h,s,d] bf16 = x @ W + bias ============
// grid (BS/16, HID/128), block 256 (8 waves, one 16x16 col-tile each)
__global__ __launch_bounds__(256) void qkv_kernel(
    const float* __restrict__ x, const float* __restrict__ W,
    const float* __restrict__ bias, bf16* __restrict__ outB) {
  __shared__ bf16 As[16*32];    // [m][k]
  __shared__ bf16 Bs[128*32];   // [n][k]  (transposed W tile)
  int tid = threadIdx.x, wave = tid >> 5, lane = tid & 31;
  int r0 = blockIdx.x * 16;
  int c0 = blockIdx.y * 128;
  v8f acc = v8f_zero();
  for (int k0 = 0; k0 < HID_; k0 += 32) {
    {  // A tile: 16x32, 2 f32 per thread (vector float2 read, v2bf store)
      int row = tid >> 4, col = (tid & 15) * 2;
      v2f xv = *(const v2f*)(x + (size_t)(r0+row)*HID_ + k0 + col);
      v2bf pk; pk[0] = f2bf(xv[0]); pk[1] = f2bf(xv[1]);
      *(v2bf*)(As + row*32 + col) = pk;
    }
#pragma unroll
    for (int t = 0; t < 4; ++t) {  // W tile: 32x128, float4 reads, transposed stores
      int e0 = (tid + t*256) * 4;
      int row = e0 >> 7, col = e0 & 127;
      v4f wv = *(const v4f*)(W + (size_t)(k0+row)*HID_ + c0 + col);
#pragma unroll
      for (int e = 0; e < 4; ++e) Bs[(col+e)*32 + row] = f2bf(wv[e]);
    }
    __syncthreads();
    v16bf af = load_a<32>(As, lane, 0);
    v16bf bfg = load_b<32>(Bs, lane, 0, wave*16 + (lane & 15));
    acc = WMMA_BF16(af, bfg, acc);
    __syncthreads();
  }
  int half = lane >> 4, nn = lane & 15;
  int o = c0 + wave*16 + nn;
  float bv = bias[o];
  int h = o >> 6, d = o & 63;
#pragma unroll
  for (int v = 0; v < 8; ++v) {
    int m = v + (half << 3);
    int row = r0 + m, bb = row >> 9, s = row & 511;
    outB[((size_t)(bb*NH_ + h)*S_ + s)*HD_ + d] = f2bf(acc[v] + bv);
  }
}

// ============ K2: qr = q @ Wr^T (bf16), qb = q . br (f32) ============
// grid ROWS/16, block 128 (4 waves, one 16-wide r-tile each)
__global__ __launch_bounds__(128) void qr_kernel(
    const bf16* __restrict__ qB, const float* __restrict__ Wr,
    const float* __restrict__ br, bf16* __restrict__ qrB, float* __restrict__ qbF) {
  __shared__ bf16 Aq[16*64];    // [m][k]
  __shared__ bf16 WrL[64*64];   // [r][d] == [n][k]
  int tid = threadIdx.x, wave = tid >> 5, lane = tid & 31;
  int r0 = blockIdx.x * 16;
  *(v8bf*)(Aq + tid*8) = *(const v8bf*)(qB + (size_t)r0*64 + tid*8);
#pragma unroll
  for (int t = 0; t < 8; ++t) {
    int e0 = (tid + t*128) * 4;
    *(v4bf*)(WrL + e0) = pack4(*(const v4f*)(Wr + e0));
  }
  __syncthreads();
  int half = lane >> 4, nn = lane & 15;
  int n = wave*16 + nn;                       // output r index
  v8f acc = v8f_zero();
#pragma unroll
  for (int k0 = 0; k0 < 64; k0 += 32) {
    v16bf af = load_a<64>(Aq, lane, k0);
    v16bf bfg = load_b<64>(WrL, lane, k0, n); // B[d][r] = Wr[r][d]
    acc = WMMA_BF16(af, bfg, acc);
  }
#pragma unroll
  for (int v = 0; v < 8; ++v) {
    int m = v + (half << 3);
    qrB[(size_t)(r0 + m)*REL_ + n] = f2bf(acc[v]);
  }
  if (tid < 16) {
    float s = 0.0f;
    for (int d = 0; d < 64; ++d) s += bf2f(Aq[tid*64 + d]) * br[d];
    qbF[r0 + tid] = s;
  }
}

// ============ K3: scores_qk[b,h,i,j] = q . k (batched per head) ============
// grid (S/16, S/128, B*NH), block 256
__global__ __launch_bounds__(256) void qk_kernel(
    const bf16* __restrict__ qB, const bf16* __restrict__ kB,
    float* __restrict__ scores) {
  __shared__ bf16 Aq[16*64];    // [i][d]
  __shared__ bf16 Kl[128*64];   // [j][d] == [n][k]
  int tid = threadIdx.x, wave = tid >> 5, lane = tid & 31;
  int bh = blockIdx.z;
  int i0 = blockIdx.x * 16;
  int j0 = blockIdx.y * 128;
  const bf16* qrow = qB + ((size_t)bh*S_ + i0)*HD_;
  const bf16* krow = kB + ((size_t)bh*S_ + j0)*HD_;
  *(v4bf*)(Aq + tid*4) = *(const v4bf*)(qrow + tid*4);
#pragma unroll
  for (int t = 0; t < 4; ++t) {
    int e0 = (tid + t*256) * 8;
    *(v8bf*)(Kl + e0) = *(const v8bf*)(krow + e0);
  }
  __syncthreads();
  int half = lane >> 4, nn = lane & 15;
  int n = wave*16 + nn;                        // local j
  v8f acc = v8f_zero();
#pragma unroll
  for (int k0 = 0; k0 < 64; k0 += 32) {
    v16bf af = load_a<64>(Aq, lane, k0);
    v16bf bfg = load_b<64>(Kl, lane, k0, n); // B[d][j] = k[j][d]
    acc = WMMA_BF16(af, bfg, acc);
  }
#pragma unroll
  for (int v = 0; v < 8; ++v) {
    int m = v + (half << 3);
    scores[((size_t)bh*S_ + i0 + m)*S_ + j0 + n] = acc[v];
  }
}

// ==== K4: per (b,i): scores += qr @ E^T (all heads), scale+mask+softmax ====
// grid (S, B), block 256. E staged in two 256-row halves (LDS <= 64KB).
__global__ __launch_bounds__(256) void rel_softmax_kernel(
    const float* __restrict__ rel, const float* __restrict__ graph,
    const float* __restrict__ endm, const bf16* __restrict__ qrB,
    const float* __restrict__ qbF, float* __restrict__ scores) {
  __shared__ bf16  El[256*64];   // 32 KB, [j][r] == [n][k]
  __shared__ bf16  Aqr[16*64];   //  2 KB
  __shared__ float Sc[12*512];   // 24 KB
  __shared__ float Msk[512];     //  2 KB
  int tid = threadIdx.x, wave = tid >> 5, lane = tid & 31;
  int i = blockIdx.x, bz = blockIdx.y;
  {  // A = qr rows (12 heads, pad to 16), 4 bf16 per thread
    int idx = tid * 4, h = idx >> 6, r = idx & 63;
    if (h < NH_) {
      *(v4bf*)(Aqr + idx) =
          *(const v4bf*)(qrB + (((size_t)bz*NH_ + h)*S_ + i)*REL_ + r);
    } else {
      v4bf z; z[0] = z[1] = z[2] = z[3] = f2bf(0.0f);
      *(v4bf*)(Aqr + idx) = z;
    }
  }
#pragma unroll
  for (int t = 0; t < 6; ++t) {  // stage qk scores rows, float4
    int e0 = (tid + t*256) * 4;
    int h = e0 >> 9, j = e0 & 511;
    *(v4f*)(Sc + e0) =
        *(const v4f*)(scores + (((size_t)bz*NH_ + h)*S_ + i)*S_ + j);
  }
  {  // mask row: (1 - (graph + end)) * (-1e9), matching reference exactly
    const float* grow = graph + ((size_t)bz*S_ + i)*S_;
    const float* erow = endm + (size_t)bz*S_;
#pragma unroll
    for (int t = 0; t < 2; ++t) {
      int j = tid + t*256;
      Msk[j] = (1.0f - (grow[j] + erow[j])) * (-1e9f);
    }
  }
  __syncthreads();
  int half = lane >> 4, nn = lane & 15;
  for (int hb = 0; hb < 2; ++hb) {
    int jbase = hb * 256;
    const float* Erow = rel + (((size_t)bz*S_ + i)*S_ + jbase)*REL_;
#pragma unroll
    for (int t = 0; t < 16; ++t) {  // 16384 elems, float4 -> v4bf
      int e0 = (tid + t*256) * 4;
      *(v4bf*)(El + e0) = pack4(*(const v4f*)(Erow + e0));
    }
    __syncthreads();
#pragma unroll
    for (int t = 0; t < 2; ++t) {
      int jloc = (wave*2 + t) * 16;
      v8f acc = v8f_zero();
#pragma unroll
      for (int k0 = 0; k0 < 64; k0 += 32) {
        v16bf af = load_a<64>(Aqr, lane, k0);
        v16bf bfg = load_b<64>(El, lane, k0, jloc + nn); // B[r][j] = E[j][r]
        acc = WMMA_BF16(af, bfg, acc);
      }
#pragma unroll
      for (int v = 0; v < 8; ++v) {
        int m = v + (half << 3);
        if (m < NH_) Sc[m*512 + jbase + jloc + nn] += acc[v];
      }
    }
    __syncthreads();
  }
  // per-head softmax over j (wave handles rows w, w+8)
  for (int h = wave; h < NH_; h += 8) {
    float qbv = qbF[((size_t)bz*NH_ + h)*S_ + i];
    float mx = -3.4e38f;
    for (int j = lane; j < 512; j += 32) {
      float val = (Sc[h*512 + j] + qbv) * 0.125f + Msk[j];
      Sc[h*512 + j] = val;
      mx = fmaxf(mx, val);
    }
#pragma unroll
    for (int off = 16; off; off >>= 1) mx = fmaxf(mx, __shfl_xor(mx, off, 32));
    float sum = 0.0f;
    for (int j = lane; j < 512; j += 32) {
      float e = __expf(Sc[h*512 + j] - mx);
      Sc[h*512 + j] = e;
      sum += e;
    }
#pragma unroll
    for (int off = 16; off; off >>= 1) sum += __shfl_xor(sum, off, 32);
    float inv = 1.0f / sum;
    float* srow = scores + (((size_t)bz*NH_ + h)*S_ + i)*S_;
    for (int j = lane; j < 512; j += 32) srow[j] = Sc[h*512 + j] * inv;
  }
}

// ============ K5: ctx = probs @ v (batched per head) -> d_out ============
// grid (S/16, B*NH), block 128 (4 waves, one 16-wide d-tile each)
__global__ __launch_bounds__(128) void ctx_kernel(
    const float* __restrict__ probs, const bf16* __restrict__ vB,
    float* __restrict__ out) {
  __shared__ bf16 Pl[16*32];   // [i][j]
  __shared__ bf16 Vl[64*32];   // [d][j]  (transposed V tile)
  int tid = threadIdx.x, wave = tid >> 5, lane = tid & 31;
  int bh = blockIdx.y, bz = bh / NH_, h = bh % NH_;
  int i0 = blockIdx.x * 16;
  const float* prow = probs + ((size_t)bh*S_ + i0)*S_;
  const bf16* vrow = vB + (size_t)bh*S_*HD_;
  int half = lane >> 4, nn = lane & 15;
  v8f acc = v8f_zero();
  for (int kk0 = 0; kk0 < S_; kk0 += 32) {
    {  // probs tile 16x32, float4 -> v4bf
      int e0 = tid * 4, r = e0 >> 5, c = e0 & 31;
      *(v4bf*)(Pl + e0) = pack4(*(const v4f*)(prow + (size_t)r*S_ + kk0 + c));
    }
#pragma unroll
    for (int t = 0; t < 2; ++t) {  // V tile 32x64: v8bf read, transposed stores
      int chunk = tid + t*128;
      int j = chunk >> 3, d0 = (chunk & 7) * 8;
      v8bf vv = *(const v8bf*)(vrow + (size_t)(kk0 + j)*HD_ + d0);
#pragma unroll
      for (int e = 0; e < 8; ++e) Vl[(d0+e)*32 + j] = vv[e];
    }
    __syncthreads();
    v16bf af = load_a<32>(Pl, lane, 0);
    v16bf bfg = load_b<32>(Vl, lane, 0, wave*16 + nn); // B[j][d] = v[j][d]
    acc = WMMA_BF16(af, bfg, acc);
    __syncthreads();
  }
#pragma unroll
  for (int v = 0; v < 8; ++v) {
    int m = v + (half << 3);
    out[((size_t)bz*S_ + i0 + m)*HID_ + h*HD_ + wave*16 + nn] = acc[v];
  }
}

extern "C" void kernel_launch(void* const* d_in, const int* in_sizes, int n_in,
                              void* d_out, int out_size, void* d_ws, size_t ws_size,
                              hipStream_t stream) {
  (void)in_sizes; (void)n_in; (void)out_size; (void)ws_size;
  const float* x     = (const float*)d_in[0];
  const float* graph = (const float*)d_in[1];
  const float* endm  = (const float*)d_in[2];
  const float* rel   = (const float*)d_in[3];
  const float* Wq = (const float*)d_in[4];  const float* bq = (const float*)d_in[5];
  const float* Wk = (const float*)d_in[6];  const float* bk = (const float*)d_in[7];
  const float* Wv = (const float*)d_in[8];  const float* bv = (const float*)d_in[9];
  const float* Wr = (const float*)d_in[10]; const float* br = (const float*)d_in[11];

  char* ws = (char*)d_ws;
  bf16*  qB  = (bf16*)(ws + OFF_Q);
  bf16*  kB  = (bf16*)(ws + OFF_K);
  bf16*  vB  = (bf16*)(ws + OFF_V);
  bf16*  qrB = (bf16*)(ws + OFF_QR);
  float* qbF = (float*)(ws + OFF_QB);
  float* sc  = (float*)(ws + OFF_SC);

  dim3 gq(BS_/16, HID_/128);
  qkv_kernel<<<gq, 256, 0, stream>>>(x, Wq, bq, qB);
  qkv_kernel<<<gq, 256, 0, stream>>>(x, Wk, bk, kB);
  qkv_kernel<<<gq, 256, 0, stream>>>(x, Wv, bv, vB);
  qr_kernel<<<ROWS_/16, 128, 0, stream>>>(qB, Wr, br, qrB, qbF);
  qk_kernel<<<dim3(S_/16, S_/128, B_*NH_), 256, 0, stream>>>(qB, kB, sc);
  rel_softmax_kernel<<<dim3(S_, B_), 256, 0, stream>>>(rel, graph, endm, qrB, qbF, sc);
  ctx_kernel<<<dim3(S_/16, B_*NH_), 128, 0, stream>>>(sc, vB, (float*)d_out);
}
